// VariationalLSTM_24223615549712
// MI455X (gfx1250) — matrix operand order repaired and checked by reading, not compiled, not run on hardware
//
#include <hip/hip_runtime.h>
#include <hip/hip_bf16.h>
#include <math.h>
#include <stdint.h>
#include <stddef.h>

// ---------------------------------------------------------------------------
// VariationalLSTM on gfx1250 (MI455X): 2 layers, B=64 T=512 D=H=256.
// Latency-bound recurrence -> 32 persistent WGs, bf16 WMMA (16x16x32),
// fused [x_t | h] @ [W_ih | W_hh]^T (K=512).
//  - 16 weight B-fragments per wave reloaded from global every timestep
//  - TWO independent 8-deep WMMA chains (x-half, h-half) interleaved for
//    ILP; summed at the end. Halves the dependent-WMMA critical path and
//    lets loads overlap the other chain's WMMAs.
//  - per-step device-scope barrier publishes h between the 32 WGs
// ---------------------------------------------------------------------------

#define B_   64
#define T_   512
#define H_   256
#define NWG  32
#define KEEP 0.75f

typedef __attribute__((ext_vector_type(16))) __bf16 v16bf;
typedef __attribute__((ext_vector_type(8)))  __bf16 v8bf;
typedef __attribute__((ext_vector_type(8)))  float  v8f;

union V16U { v16bf v; v8bf h[2]; };

// A fragment (16x32 bf16, M = lane&15):
//   lanes 0-15 : K = kb+{0..7}  and kb+{16..23}
//   lanes16-31 : K = kb+{8..15} and kb+{24..31}
__device__ inline v16bf load_a(const __bf16* rowp, int kb, int lane) {
  const __bf16* p = rowp + kb + ((lane >> 4) << 3);
  V16U r;
  r.h[0] = *(const v8bf*)(p);
  r.h[1] = *(const v8bf*)(p + 16);
  return r.v;
}

// B fragment (32x16 bf16, N = lane&15):
//   lanes 0-15 : K = kb+{0..7} and kb+{16..23}, lanes16-31 : K = kb+{8..15} and kb+{24..31}
__device__ inline v16bf load_b(const __bf16* rowp, int kb, int lane) {
  const __bf16* p = rowp + kb + ((lane >> 4) << 3);
  V16U r;
  r.h[0] = *(const v8bf*)(p);
  r.h[1] = *(const v8bf*)(p + 16);
  return r.v;
}

__device__ inline float sigmoidf_(float x) { return 1.0f / (1.0f + __expf(-x)); }

// ---------------------------------------------------------------------------
// Persistent per-layer LSTM kernel: grid = 32 WGs x 256 threads (8 wave32).
// WG g: batch chunk bc = g&3 (16 rows), hidden slice hs = g>>2 (32 units).
// Wave w owns one 16x16 output tile: gate = w>>1, j = w&1,
//   global N base = gate*256 + hs*32 + j*16.
// ---------------------------------------------------------------------------
__global__ __launch_bounds__(256)
void lstm_layer_kernel(const __bf16* __restrict__ Xin,   // [B,T,256] bf16
                       const __bf16* __restrict__ Wih,   // [1024,256] bf16
                       const __bf16* __restrict__ Whh,   // [1024,256] bf16
                       const float*  __restrict__ b_ih,  // [1024]
                       const float*  __restrict__ b_hh,  // [1024]
                       const float*  __restrict__ m_out, // [B,H]
                       const float*  __restrict__ m_h,   // [B,H]
                       const float*  __restrict__ m_c,   // [B,H]
                       __bf16*       __restrict__ hpub,  // [B,H] bf16 (zeroed)
                       unsigned*     __restrict__ cells, // [T] (zeroed)
                       float*        __restrict__ Yf,    // [B,T,H] f32 out
                       __bf16*       __restrict__ Ybf,   // [B,T,H] bf16 out
                       int storeF32)
{
  __shared__ float gates[16 * 128];   // [batch in chunk][4 gates * 32 units]
  __shared__ float cst[16 * 32];      // persistent cell state (masked)

  const int tid  = threadIdx.x;
  const int lane = tid & 31;
  const int wave = tid >> 5;
  const int wg   = blockIdx.x;
  const int bc   = wg & 3;            // batch chunk (16 rows)
  const int hs   = wg >> 2;           // hidden slice (32 units)

  const int rowA = bc * 16 + (lane & 15);          // batch row for A frags

  // One 16x16 output tile per wave.
  const int gate = wave >> 1;
  const int jj   = wave & 1;
  const int N0   = gate * 256 + hs * 32 + jj * 16;
  const __bf16* wih0 = Wih + (size_t)(N0 + (lane & 15)) * 256;
  const __bf16* whh0 = Whh + (size_t)(N0 + (lane & 15)) * 256;

  // Per-thread elementwise preloads (2 (b,u) pairs per thread).
  float bi[2], bff[2], bgg[2], bo[2], mo[2], mh[2], mc[2];
#pragma unroll
  for (int q = 0; q < 2; ++q) {
    int pi = tid + 256 * q;
    int b = pi & 15, u = pi >> 4;       // u in [0,32)
    int ug = hs * 32 + u;
    int bg = bc * 16 + b;
    bi[q]  = b_ih[0 * 256 + ug] + b_hh[0 * 256 + ug];
    bff[q] = b_ih[1 * 256 + ug] + b_hh[1 * 256 + ug];
    bgg[q] = b_ih[2 * 256 + ug] + b_hh[2 * 256 + ug];
    bo[q]  = b_ih[3 * 256 + ug] + b_hh[3 * 256 + ug];
    mo[q]  = m_out[bg * 256 + ug];
    mh[q]  = m_h[bg * 256 + ug];
    mc[q]  = m_c[bg * 256 + ug];
    cst[b * 32 + u] = 0.0f;
  }
  __syncthreads();

  const __bf16* hrow = hpub + (size_t)rowA * 256;

  for (int t = 0; t < T_; ++t) {
    const __bf16* xrow = Xin + ((size_t)rowA * T_ + t) * 256;
    // prefetch next timestep's x row (global_prefetch path)
    __builtin_prefetch(xrow + 256, 0, 1);

    // Two independent accumulation chains (x-half and h-half), interleaved:
    // each chain's loads can issue under the other chain's WMMA execution,
    // and the dependent-WMMA depth per chain is 8 instead of 16.
    v8f accx = {};
    v8f acch = {};
#pragma unroll
    for (int ks = 0; ks < 8; ++ks) {
      v16bf ax = load_a(xrow, ks * 32, lane);
      v16bf bx = load_b(wih0, ks * 32, lane);
      v16bf ah = load_a(hrow, ks * 32, lane);
      v16bf bh = load_b(whh0, ks * 32, lane);
      accx = __builtin_amdgcn_wmma_f32_16x16x32_bf16(false, ax, false, bx, (short)0, accx, false, false);
      acch = __builtin_amdgcn_wmma_f32_16x16x32_bf16(false, ah, false, bh, (short)0, acch, false, false);
    }
    v8f acc0 = accx + acch;

    // D layout: VGPR r -> M = r + 8*(lane>=16), N = lane&15
    {
      int nl = wave * 16 + (lane & 15);     // = gate*32 + jj*16 + (lane&15)
      int mr = (lane >> 4) * 8;
#pragma unroll
      for (int r = 0; r < 8; ++r)
        gates[(mr + r) * 128 + nl] = acc0[r];
    }
    __syncthreads();

    // Elementwise LSTM cell update + variational dropout
#pragma unroll
    for (int q = 0; q < 2; ++q) {
      int pi = tid + 256 * q;
      int b = pi & 15, u = pi >> 4;       // u in [0,32)
      float iv = gates[b * 128 + 0  + u] + bi[q];
      float fv = gates[b * 128 + 32 + u] + bff[q];
      float gv = gates[b * 128 + 64 + u] + bgg[q];
      float ov = gates[b * 128 + 96 + u] + bo[q];
      float is = sigmoidf_(iv);
      float fs = sigmoidf_(fv);
      float os = sigmoidf_(ov);
      float cn = fs * cst[b * 32 + u] + is * tanhf(gv);
      float hn = os * tanhf(cn);
      cst[b * 32 + u] = cn * mc[q];
      int bg = bc * 16 + b;
      int ug = hs * 32 + u;
      hpub[bg * 256 + ug] = (__bf16)(hn * mh[q]);
      size_t oidx = ((size_t)bg * T_ + t) * 256 + ug;
      float out = hn * mo[q];
      if (storeF32) Yf[oidx] = out;
      else          Ybf[oidx] = (__bf16)out;
    }

    // Device barrier across the 32 persistent WGs (publish h).
    __syncthreads();
    if (tid == 0) {
      __threadfence();
      __hip_atomic_fetch_add(&cells[t], 1u, __ATOMIC_ACQ_REL, __HIP_MEMORY_SCOPE_AGENT);
      while (__hip_atomic_load(&cells[t], __ATOMIC_ACQUIRE, __HIP_MEMORY_SCOPE_AGENT) < (unsigned)NWG)
        __builtin_amdgcn_s_sleep(2);
    }
    __syncthreads();
    // acquire in every wave so stale h is not served from near caches
    (void)__hip_atomic_load(&cells[t], __ATOMIC_ACQUIRE, __HIP_MEMORY_SCOPE_AGENT);
  }
}

// ---------------------------------------------------------------------------
// Threefry2x32-20 (JAX-compatible) for the variational dropout masks.
// ---------------------------------------------------------------------------
__device__ inline uint32_t rotl32_(uint32_t x, uint32_t n) { return (x << n) | (x >> (32u - n)); }

__device__ inline void threefry_(uint32_t k0, uint32_t k1, uint32_t c0, uint32_t c1,
                                 uint32_t* o0, uint32_t* o1) {
  uint32_t ksv[3] = {k0, k1, k0 ^ k1 ^ 0x1BD11BDAu};
  uint32_t x0 = c0 + ksv[0], x1 = c1 + ksv[1];
  const uint32_t R[2][4] = {{13u, 15u, 26u, 6u}, {17u, 29u, 16u, 24u}};
#pragma unroll
  for (int i = 0; i < 5; ++i) {
#pragma unroll
    for (int r = 0; r < 4; ++r) {
      x0 += x1; x1 = rotl32_(x1, R[i & 1][r]); x1 ^= x0;
    }
    x0 += ksv[(i + 1) % 3];
    x1 += ksv[(i + 2) % 3] + (uint32_t)(i + 1);
  }
  *o0 = x0; *o1 = x1;
}

__device__ inline void split2_(uint32_t k0, uint32_t k1, uint32_t out[2][2]) {
  uint32_t a0, b0, a1, b1;
  threefry_(k0, k1, 0u, 2u, &a0, &b0);
  threefry_(k0, k1, 1u, 3u, &a1, &b1);
  out[0][0] = a0; out[0][1] = a1;   // flat [a0,a1,b0,b1] reshaped (2,2)
  out[1][0] = b0; out[1][1] = b1;
}

__device__ inline void split3_(uint32_t k0, uint32_t k1, uint32_t out[3][2]) {
  uint32_t A0, B0, A1, B1, A2, B2;
  threefry_(k0, k1, 0u, 3u, &A0, &B0);
  threefry_(k0, k1, 1u, 4u, &A1, &B1);
  threefry_(k0, k1, 2u, 5u, &A2, &B2);
  out[0][0] = A0; out[0][1] = A1;   // flat [A0,A1,A2,B0,B1,B2] reshaped (3,2)
  out[1][0] = A2; out[1][1] = B0;
  out[2][0] = B1; out[2][1] = B2;
}

__device__ inline float bern_(uint32_t bits) {
  uint32_t f = (bits >> 9) | 0x3f800000u;
  float u = __uint_as_float(f) - 1.0f;
  return (u < KEEP) ? (1.0f / KEEP) : 0.0f;
}

__global__ void mask_kernel(float* __restrict__ masks) {  // 6 masks x 16384 floats
  int tid = blockIdx.x * blockDim.x + threadIdx.x;
  if (tid >= 6 * 8192) return;
  int midx = tid / 8192;        // layer*3 + {out,h,c}
  int i    = tid % 8192;
  int layer = midx / 3, m = midx % 3;

  uint32_t lk[2][2]; split2_(0u, 42u, lk);                 // MASK_KEY = key(42)
  uint32_t mk[3][2]; split3_(lk[layer][0], lk[layer][1], mk);
  uint32_t A, Bv;
  threefry_(mk[m][0], mk[m][1], (uint32_t)i, (uint32_t)(8192 + i), &A, &Bv);

  float* out = masks + (size_t)midx * 16384;
  out[i]        = bern_(A);
  out[8192 + i] = bern_(Bv);
}

// ---------------------------------------------------------------------------
__global__ void f32_to_bf16_kernel(const float* __restrict__ in, __bf16* __restrict__ out, int n) {
  int i = blockIdx.x * blockDim.x + threadIdx.x;
  if (i < n) out[i] = (__bf16)in[i];
}

__global__ void zero_u32_kernel(uint32_t* __restrict__ p, int n) {
  int i = blockIdx.x * blockDim.x + threadIdx.x;
  if (i < n) p[i] = 0u;
}

// ---------------------------------------------------------------------------
extern "C" void kernel_launch(void* const* d_in, const int* in_sizes, int n_in,
                              void* d_out, int out_size, void* d_ws, size_t ws_size,
                              hipStream_t stream) {
  (void)in_sizes; (void)n_in; (void)out_size; (void)ws_size;

  const float* x    = (const float*)d_in[0];
  const float* Wih0 = (const float*)d_in[1];
  const float* Whh0 = (const float*)d_in[2];
  const float* bih0 = (const float*)d_in[3];
  const float* bhh0 = (const float*)d_in[4];
  const float* Wih1 = (const float*)d_in[5];
  const float* Whh1 = (const float*)d_in[6];
  const float* bih1 = (const float*)d_in[7];
  const float* bhh1 = (const float*)d_in[8];

  char* ws = (char*)d_ws;
  size_t off = 0;
  auto alloc = [&](size_t bytes) -> void* {
    void* p = ws + off;
    off = (off + bytes + 255) & ~(size_t)255;
    return p;
  };

  const int BTD = B_ * T_ * 256;                      // 8388608
  __bf16* X0    = (__bf16*)alloc((size_t)BTD * 2);    // layer0 input bf16
  __bf16* X1    = (__bf16*)alloc((size_t)BTD * 2);    // layer0 output / layer1 input bf16
  __bf16* Wih0b = (__bf16*)alloc(1024 * 256 * 2);
  __bf16* Whh0b = (__bf16*)alloc(1024 * 256 * 2);
  __bf16* Wih1b = (__bf16*)alloc(1024 * 256 * 2);
  __bf16* Whh1b = (__bf16*)alloc(1024 * 256 * 2);
  float*  masks = (float*)alloc(6 * 16384 * 4);
  __bf16* hpub  = (__bf16*)alloc(B_ * H_ * 2);        // 32768 B (256-aligned)
  unsigned* cells = (unsigned*)alloc(2 * T_ * 4);     // contiguous after hpub

  // bf16 conversions
  f32_to_bf16_kernel<<<(BTD + 255) / 256, 256, 0, stream>>>(x, X0, BTD);
  f32_to_bf16_kernel<<<1024, 256, 0, stream>>>(Wih0, Wih0b, 1024 * 256);
  f32_to_bf16_kernel<<<1024, 256, 0, stream>>>(Whh0, Whh0b, 1024 * 256);
  f32_to_bf16_kernel<<<1024, 256, 0, stream>>>(Wih1, Wih1b, 1024 * 256);
  f32_to_bf16_kernel<<<1024, 256, 0, stream>>>(Whh1, Whh1b, 1024 * 256);

  // dropout masks (threefry)
  mask_kernel<<<192, 256, 0, stream>>>(masks);

  // zero h and both layers' barrier cells (hpub and cells are contiguous)
  zero_u32_kernel<<<((B_ * H_ * 2 + 2 * T_ * 4) / 4 + 255) / 256, 256, 0, stream>>>(
      (uint32_t*)hpub, (B_ * H_ * 2 + 2 * T_ * 4) / 4);

  // layer 0: bf16 output into X1
  lstm_layer_kernel<<<NWG, 256, 0, stream>>>(
      X0, Wih0b, Whh0b, bih0, bhh0,
      masks + 0 * 16384, masks + 1 * 16384, masks + 2 * 16384,
      hpub, cells, (float*)d_out, X1, /*storeF32=*/0);

  // reset h between layers
  zero_u32_kernel<<<(B_ * H_ * 2 / 4 + 255) / 256, 256, 0, stream>>>(
      (uint32_t*)hpub, B_ * H_ * 2 / 4);

  // layer 1: fp32 output into d_out
  lstm_layer_kernel<<<NWG, 256, 0, stream>>>(
      X1, Wih1b, Whh1b, bih1, bhh1,
      masks + 3 * 16384, masks + 4 * 16384, masks + 5 * 16384,
      hpub, cells + T_, (float*)d_out, X1, /*storeF32=*/1);
}
